// Mvdr_67104569033323
// MI455X (gfx1250) — matrix operand order repaired and checked by reading, not compiled, not run on hardware
//
#include <hip/hip_runtime.h>
#include <hip/hip_bf16.h>
#include <math.h>

// ---------------- problem constants ----------------
constexpr int kNFFT  = 512;
constexpr int kWIN   = 320;
constexpr int kHOP   = 160;
constexpr int kPAD   = 256;              // NFFT/2
constexpr int kF     = 257;              // NFFT/2+1
constexpr int kDF    = 514;              // 2*F
constexpr int kKF    = 544;              // DF padded to multiple of 32
constexpr int kT     = 601;              // frames
constexpr int kB     = 8;
constexpr int kC     = 8;
constexpr int kNSEQ  = kB * kC;          // 64
constexpr int kL     = 96000;
constexpr int kH     = 128;
constexpr int kG     = 512;              // 4*H gates
constexpr int kNPAD  = 528;              // 514 padded to multiple of 16
constexpr int kM     = kT * kNSEQ;       // 38464 (divisible by 16)
constexpr int kMI    = 4816;             // B*T=4808 padded to multiple of 16
constexpr int kTOTAL = (kT - 1) * kHOP + kNFFT;  // 96512
constexpr float kTWO_PI = 6.28318530717958647692f;

typedef _Float16 h16;
typedef __attribute__((ext_vector_type(16))) _Float16 v16h;
typedef __attribute__((ext_vector_type(8)))  _Float16 v8h;
typedef __attribute__((ext_vector_type(8)))  float    v8f;
typedef __attribute__((ext_vector_type(4)))  unsigned int u32x4;
typedef __attribute__((ext_vector_type(8)))  int i32x8;
typedef __attribute__((ext_vector_type(4)))  int i32x4;

union V16H { v16h v; v8h h[2]; };

#if defined(__has_builtin)
#if __has_builtin(__builtin_amdgcn_tensor_load_to_lds) && __has_builtin(__builtin_amdgcn_s_wait_tensorcnt)
#define USE_TDM 1
#endif
#endif

// ---------------- WMMA fragment helpers (CDNA5 ISA 7.12.2 layouts) ----------------
// A (16x32 f16, M x K): lane&15 = row; lane>>4 selects K-halves {0..7,16..23} vs {8..15,24..31}
__device__ __forceinline__ v16h load_a16(const h16* A, int lda, int row0, int k0, int lane) {
  int r  = row0 + (lane & 15);
  int kh = (lane >> 4) << 3;            // 0 or 8
  V16H u;
  u.h[0] = *(const v8h*)(A + (size_t)r * lda + k0 + kh);
  u.h[1] = *(const v8h*)(A + (size_t)r * lda + k0 + 16 + kh);
  return u.v;
}
// B (32x16 f16, K x N) with B[k][n] = Wt[n][k] (N-major storage, contiguous K)
__device__ __forceinline__ v16h load_b16(const h16* Wt, int ldb, int n0, int k0, int lane) {
  int n  = n0 + (lane & 15);
  int kb = (lane >> 4) << 3;            // 0 or 8
  const h16* p = Wt + (size_t)n * ldb + k0 + kb;
  V16H u;
  u.h[0] = *(const v8h*)(p);
  u.h[1] = *(const v8h*)(p + 16);
  return u.v;
}

__device__ __forceinline__ float hann_win(int k) {
  float w = 0.f;
  if (k >= 96 && k < 416)
    w = 0.5f - 0.5f * cosf(kTWO_PI * (float)(k - 96) / (float)kWIN);
  return w;
}

// ---------------- weight prep: f32 -> f16, pad, fuse biases ----------------
__global__ void prep_weights(const float* Wih, const float* Whh, const float* bih,
                             const float* bhh, const float* W1, const float* W2,
                             h16* Wih_h, h16* Whh_h, h16* W1h, h16* W2h, float* bias) {
  int idx = blockIdx.x * blockDim.x + threadIdx.x;
  int stride = gridDim.x * blockDim.x;
  for (int i = idx; i < kG * kKF; i += stride) {
    int n = i / kKF, k = i - n * kKF;
    Wih_h[i] = (k < kDF) ? (h16)Wih[(size_t)n * kDF + k] : (h16)0.f;
  }
  for (int i = idx; i < kG * kH; i += stride) Whh_h[i] = (h16)Whh[i];
  for (int i = idx; i < kNPAD * kH; i += stride) {
    int j = i / kH, k = i - j * kH;
    W1h[i] = (j < kDF) ? (h16)W1[(size_t)j * kH + k] : (h16)0.f;
    W2h[i] = (j < kDF) ? (h16)W2[(size_t)j * kH + k] : (h16)0.f;
  }
  for (int i = idx; i < kG; i += stride) bias[i] = bih[i] + bhh[i];
}

// ---------------- DFT / iDFT basis matrices (f16, N-major with contiguous K) ----------------
// dftB [528 rows n][512 k]:  n<257: cos(2*pi*n*k/512); 257<=n<514: -sin(2*pi*(n-257)*k/512)
// idftB[512 rows k][544 n]:  n<257:  w_f/512 * cos(2*pi*n*k/512)
//                            257<=n<514: -w_f/512 * sin(2*pi*(n-257)*k/512),  w_f = (f==0||f==256)?1:2
__global__ void prep_basis(h16* dftB, h16* idftB) {
  int idx = blockIdx.x * blockDim.x + threadIdx.x;
  int stride = gridDim.x * blockDim.x;
  for (int i = idx; i < kNPAD * kNFFT; i += stride) {
    int n = i / kNFFT, k = i - n * kNFFT;
    float v = 0.f;
    if (n < kF) {
      int j = (n * k) & (kNFFT - 1);
      v = cosf(kTWO_PI * (float)j / (float)kNFFT);
    } else if (n < kDF) {
      int j = ((n - kF) * k) & (kNFFT - 1);
      v = -sinf(kTWO_PI * (float)j / (float)kNFFT);
    }
    dftB[i] = (h16)v;
  }
  for (int i = idx; i < kNFFT * kKF; i += stride) {
    int k = i / kKF, n = i - k * kKF;
    float v = 0.f;
    if (n < kF) {
      float wf = (n == 0 || n == 256) ? 1.f : 2.f;
      int j = (n * k) & (kNFFT - 1);
      v = wf * (1.f / (float)kNFFT) * cosf(kTWO_PI * (float)j / (float)kNFFT);
    } else if (n < kDF) {
      int f = n - kF;
      float wf = (f == 0 || f == 256) ? 1.f : 2.f;
      int j = (f * k) & (kNFFT - 1);
      v = -wf * (1.f / (float)kNFFT) * sinf(kTWO_PI * (float)j / (float)kNFFT);
    }
    idftB[i] = (h16)v;
  }
}

// ---------------- windowed frames (reflect pad), f16 ----------------
__global__ void frames_kernel(const float* __restrict__ x, h16* __restrict__ frames_h) {
  int t = blockIdx.x, n = blockIdx.y;
  int b = n >> 3, c = n & 7;
  int tid = threadIdx.x;
  size_t r = (size_t)t * kNSEQ + n;
  for (int k = tid; k < kNFFT; k += blockDim.x) {
    int src = t * kHOP + k - kPAD;
    if (src < 0) src = -src;
    if (src >= kL) src = 2 * kL - 2 - src;
    frames_h[r * kNFFT + k] = (h16)(x[((size_t)b * kL + src) * kC + c] * hann_win(k));
  }
}

// ---------------- STFT as WMMA GEMM: spec = frames @ dftB^T  [38464 x 514] ----------------
__global__ void gemm_spec(const h16* __restrict__ frames, const h16* __restrict__ dftB,
                          float* __restrict__ spec_ri, h16* __restrict__ feat_h) {
  int lane = threadIdx.x & 31;
  int wave = threadIdx.x >> 5;
  int mt = blockIdx.x;
  int nt = blockIdx.y * 8 + wave;
  if (nt >= kNPAD / 16) return;                    // 33 N-tiles
  int row0 = mt * 16, col0 = nt * 16;
  v8f acc = {};
#pragma unroll
  for (int k0 = 0; k0 < kNFFT; k0 += 32) {
    v16h a = load_a16(frames, kNFFT, row0, k0, lane);
    v16h b = load_b16(dftB, kNFFT, col0, k0, lane);
    acc = __builtin_amdgcn_wmma_f32_16x16x32_f16(false, a, false, b, (short)0, acc, false, false);
  }
  int col = col0 + (lane & 15);
  if (col < kDF) {
    int rb = row0 + ((lane >> 4) << 3);
#pragma unroll
    for (int j = 0; j < 8; ++j) {
      float v = acc[j];
      spec_ri[(size_t)(rb + j) * kDF + col] = v;
      feat_h[(size_t)(rb + j) * kKF + col] = (h16)v;
    }
  }
}

// zero the K-pad columns [514,544) of feat_h
__global__ void feat_pad(h16* __restrict__ feat_h) {
  size_t idx = (size_t)blockIdx.x * blockDim.x + threadIdx.x;
  size_t total = (size_t)kM * (kKF - kDF);
  if (idx >= total) return;
  size_t r = idx / (kKF - kDF);
  int cc = (int)(idx - r * (kKF - kDF));
  feat_h[r * kKF + kDF + cc] = (h16)0.f;
}

// ---------------- GEMM: xW = feat @ W_ih^T + (b_ih+b_hh)  [38464 x 512] ----------------
__global__ void gemm_xw(const h16* __restrict__ A, const h16* __restrict__ Bw,
                        const float* __restrict__ bias, float* __restrict__ Cout) {
  int lane = threadIdx.x & 31;
  int wave = threadIdx.x >> 5;
  int mt = blockIdx.x;
  int nt = blockIdx.y * 8 + wave;                  // 0..31
  int row0 = mt * 16, col0 = nt * 16;
  v8f acc = {};
  for (int k0 = 0; k0 < kKF; k0 += 32) {
    if (k0 + 32 < kKF)
      __builtin_prefetch(A + (size_t)(row0 + (lane & 15)) * kKF + k0 + 32, 0, 1);
    v16h a = load_a16(A, kKF, row0, k0, lane);
    v16h b = load_b16(Bw, kKF, col0, k0, lane);
    acc = __builtin_amdgcn_wmma_f32_16x16x32_f16(false, a, false, b, (short)0, acc, false, false);
  }
  int col = col0 + (lane & 15);
  float bv = bias[col];
  int rb = row0 + ((lane >> 4) << 3);
#pragma unroll
  for (int j = 0; j < 8; ++j)
    Cout[(size_t)(rb + j) * kG + col] = acc[j] + bv;
}

// ---------------- LSTM: one WGP, W_hh via TDM into LDS, WMMA recurrence ----------------
__global__ void __launch_bounds__(1024) lstm_kernel(const float* __restrict__ xW,
                                                    const h16* __restrict__ Whh_h,
                                                    h16* __restrict__ h_all) {
  extern __shared__ char smem[];
  h16*   sWhh = (h16*)smem;                                  // 512*128 f16 = 128 KB (LDS offset 0)
  h16*   sH   = (h16*)(smem + 131072);                       // 64*128 f16  =  16 KB
  float* sC   = (float*)(smem + 131072 + 16384);             // 64*128 f32  =  32 KB
  float* sG   = (float*)(smem + 131072 + 16384 + 32768);     // 64*512 f32  = 128 KB
  int tid = threadIdx.x;
  int lane = tid & 31, wave = tid >> 5;                      // 32 waves

#ifdef USE_TDM
  // Tensor Data Mover: one 2D descriptor copies 128KB W_hh -> LDS (16384 x 8B elements)
  if (wave == 0) {
    unsigned long long ga = (unsigned long long)(const void*)Whh_h;
    u32x4 g0;
    g0[0] = 1u;                                       // count=1, user mode, no gather
    g0[1] = 0u;                                       // lds_addr = 0 (sWhh at LDS base)
    g0[2] = (unsigned)(ga & 0xffffffffu);             // global_addr[31:0]   (bits 95:64)
    g0[3] = (unsigned)((ga >> 32) & 0x01ffffffu)      // global_addr[56:32]  (bits 120:96)
          | (2u << 30);                               // type = 2 ("image")
    i32x8 g1;
    g1[0] = (int)(3u << 16);                          // workgroup_mask=0, data_size=3 (8B)
    g1[1] = (int)(16384u << 16);                      // tensor_dim0[15:0] << 16
    g1[2] = (int)((16384u >> 16) | (1u << 16));       // tensor_dim0[31:16] | tensor_dim1=1
    g1[3] = (int)(16384u << 16);                      // tile_dim0 = 16384
    g1[4] = 1;                                        // tile_dim1 = 1
    g1[5] = 16384;                                    // tensor_dim0_stride[31:0]
    g1[6] = 0;
    g1[7] = 0;
    i32x4 z4 = {0, 0, 0, 0};
    i32x8 z8 = {0, 0, 0, 0, 0, 0, 0, 0};
    // clang-23 / therock-10.0 lane: 6-arg form (g0, g1, g2, g3, pad, cpol)
    __builtin_amdgcn_tensor_load_to_lds(g0, g1, z4, z4, z8, 0);
  }
#else
  {
    const uint4* src = (const uint4*)Whh_h;
    uint4* dst = (uint4*)sWhh;
    for (int i = tid; i < (kG * kH) / 8; i += 1024) dst[i] = src[i];
  }
#endif
  for (int i = tid; i < kNSEQ * kH; i += 1024) { sH[i] = (h16)0.f; sC[i] = 0.f; }
#ifdef USE_TDM
  if (wave == 0) __builtin_amdgcn_s_wait_tensorcnt(0);
#endif
  __syncthreads();

  for (int t = 0; t < kT; ++t) {
    // g = xW[t] + h_{t-1} @ W_hh^T   (M=64, N=512, K=128) -> 128 tiles / 32 waves
    int nt = wave;
#pragma unroll
    for (int mt = 0; mt < 4; ++mt) {
      int row0 = mt * 16, col0 = nt * 16;
      int col = col0 + (lane & 15);
      int rb  = row0 + ((lane >> 4) << 3);
      v8f acc;
#pragma unroll
      for (int j = 0; j < 8; ++j)
        acc[j] = xW[((size_t)t * kNSEQ + rb + j) * kG + col];
#pragma unroll
      for (int k0 = 0; k0 < kH; k0 += 32) {
        v16h a = load_a16(sH, kH, row0, k0, lane);
        v16h b = load_b16(sWhh, kH, col0, k0, lane);
        acc = __builtin_amdgcn_wmma_f32_16x16x32_f16(false, a, false, b, (short)0, acc, false, false);
      }
#pragma unroll
      for (int j = 0; j < 8; ++j)
        sG[(rb + j) * kG + col] = acc[j];
    }
    __syncthreads();
    // gates (torch order i,f,g,o): 8192 entries / 1024 threads
#pragma unroll
    for (int i = 0; i < 8; ++i) {
      int e = i * 1024 + tid;
      int row = e >> 7, hh = e & 127;
      float gi = sG[row * kG + hh];
      float gf = sG[row * kG + 128 + hh];
      float gg = sG[row * kG + 256 + hh];
      float go = sG[row * kG + 384 + hh];
      float cv = sC[e];
      float si = 1.f / (1.f + __expf(-gi));
      float sf = 1.f / (1.f + __expf(-gf));
      float so = 1.f / (1.f + __expf(-go));
      cv = sf * cv + si * tanhf(gg);
      float hv = so * tanhf(cv);
      sC[e] = cv;
      sH[e] = (h16)hv;
      h_all[((size_t)t * kNSEQ + row) * kH + hh] = (h16)hv;
    }
    __syncthreads();
  }
}

// ---------------- GEMM: mask = h_all @ W^T + b  [38464 x 514] ----------------
__global__ void gemm_mask(const h16* __restrict__ Hin, const h16* __restrict__ Wm,
                          const float* __restrict__ bvec, float* __restrict__ Out) {
  int lane = threadIdx.x & 31;
  int wave = threadIdx.x >> 5;
  int mt = blockIdx.x;
  int nt = blockIdx.y * 8 + wave;
  if (nt >= kNPAD / 16) return;                    // 33 N-tiles
  int row0 = mt * 16, col0 = nt * 16;
  v8f acc = {};
#pragma unroll
  for (int k0 = 0; k0 < kH; k0 += 32) {
    v16h a = load_a16(Hin, kH, row0, k0, lane);
    v16h b = load_b16(Wm, kH, col0, k0, lane);
    acc = __builtin_amdgcn_wmma_f32_16x16x32_f16(false, a, false, b, (short)0, acc, false, false);
  }
  int col = col0 + (lane & 15);
  if (col < kDF) {
    float bv = bvec[col];
    int rb = row0 + ((lane >> 4) << 3);
#pragma unroll
    for (int j = 0; j < 8; ++j)
      Out[(size_t)(rb + j) * kDF + col] = acc[j] + bv;
  }
}

// ---------------- complex mask multiply (in place on mask buffers) ----------------
__global__ void mask_apply(float* __restrict__ s_ri, float* __restrict__ n_ri,
                           const float* __restrict__ spec_ri) {
  size_t idx = (size_t)blockIdx.x * blockDim.x + threadIdx.x;
  size_t total = (size_t)kM * kF;
  if (idx >= total) return;
  size_t r = idx / kF;
  int f = (int)(idx - r * kF);
  size_t i0 = r * kDF + f, i1 = r * kDF + kF + f;
  float re = spec_ri[i0], ie = spec_ri[i1];
  float rm = s_ri[i0], im = s_ri[i1];
  s_ri[i0] = rm * re - im * ie;
  s_ri[i1] = rm * ie + im * re;
  rm = n_ri[i0]; im = n_ri[i1];
  n_ri[i0] = rm * re - im * ie;
  n_ri[i1] = rm * ie + im * re;
}

// ---------------- per-(b,f) 8x8 complex PSD accumulation + diagonal loading ----------------
__global__ void psd_kernel(const float* __restrict__ s_ri, const float* __restrict__ n_ri,
                           float2* __restrict__ psd_s, float2* __restrict__ psd_n) {
  __shared__ float sSr[8][32], sSi[8][32], sNr[8][32], sNi[8][32];
  __shared__ float sTr[8];
  int f = blockIdx.x, b = blockIdx.y;
  int tid = threadIdx.x;                            // 64 threads
  int c = tid >> 3, e = tid & 7;
  int lch = tid & 7, lbase = tid >> 3;
  float asr = 0, asi = 0, anr = 0, ani = 0;
  for (int tc = 0; tc < kT; tc += 32) {
#pragma unroll
    for (int i = 0; i < 4; ++i) {
      int lt = lbase + 8 * i;
      int tt = tc + lt;
      float vsr = 0, vsi = 0, vnr = 0, vni = 0;
      if (tt < kT) {
        size_t r = (size_t)tt * kNSEQ + b * 8 + lch;
        vsr = s_ri[r * kDF + f]; vsi = s_ri[r * kDF + kF + f];
        vnr = n_ri[r * kDF + f]; vni = n_ri[r * kDF + kF + f];
      }
      sSr[lch][lt] = vsr; sSi[lch][lt] = vsi;
      sNr[lch][lt] = vnr; sNi[lch][lt] = vni;
    }
    __syncthreads();
#pragma unroll 4
    for (int lt = 0; lt < 32; ++lt) {
      float ar = sSr[c][lt], ai = sSi[c][lt], br = sSr[e][lt], bi = sSi[e][lt];
      asr += ar * br + ai * bi;                     // s_c * conj(s_e)
      asi += ai * br - ar * bi;
      ar = sNr[c][lt]; ai = sNi[c][lt]; br = sNr[e][lt]; bi = sNi[e][lt];
      anr += ar * br + ai * bi;
      ani += ai * br - ar * bi;
    }
    __syncthreads();
  }
  float invT = 1.0f / (float)kT;
  asr *= invT; asi *= invT; anr *= invT; ani *= invT;
  if (c == e) sTr[c] = anr;
  __syncthreads();
  float tr = 0.f;
#pragma unroll
  for (int i = 0; i < 8; ++i) tr += sTr[i];
  if (c == e) anr += 1e-6f * tr / 8.0f + 1e-8f;     // diagonal loading
  size_t o = ((size_t)b * kF + f) * 64 + tid;
  psd_s[o] = make_float2(asr, asi);
  psd_n[o] = make_float2(anr, ani);
}

// ---------------- 8x8 complex Gauss-Jordan solve + MVDR weight ----------------
__device__ __forceinline__ float2 cmul(float2 a, float2 b) {
  return make_float2(a.x * b.x - a.y * b.y, a.x * b.y + a.y * b.x);
}
__device__ __forceinline__ float2 csub(float2 a, float2 b) {
  return make_float2(a.x - b.x, a.y - b.y);
}
__device__ __forceinline__ float2 cdiv(float2 a, float2 b) {
  float d = b.x * b.x + b.y * b.y + 1e-30f;
  return make_float2((a.x * b.x + a.y * b.y) / d, (a.y * b.x - a.x * b.y) / d);
}

__global__ void solve_kernel(const float2* __restrict__ psd_s, const float2* __restrict__ psd_n,
                             float2* __restrict__ wvec) {
  __shared__ float2 As[32][64];
  __shared__ float2 Xs[32][64];
  int tid = threadIdx.x;
  int sys = blockIdx.x * 32 + tid;
  if (sys >= kB * kF) return;
  float2* A = As[tid];
  float2* X = Xs[tid];
  for (int i = 0; i < 64; ++i) {
    A[i] = psd_n[(size_t)sys * 64 + i];
    X[i] = psd_s[(size_t)sys * 64 + i];
  }
  for (int k = 0; k < 8; ++k) {
    int p = k;
    float best = A[k * 8 + k].x * A[k * 8 + k].x + A[k * 8 + k].y * A[k * 8 + k].y;
    for (int i = k + 1; i < 8; ++i) {
      float m = A[i * 8 + k].x * A[i * 8 + k].x + A[i * 8 + k].y * A[i * 8 + k].y;
      if (m > best) { best = m; p = i; }
    }
    if (p != k)
      for (int j = 0; j < 8; ++j) {
        float2 tA = A[k * 8 + j]; A[k * 8 + j] = A[p * 8 + j]; A[p * 8 + j] = tA;
        float2 tX = X[k * 8 + j]; X[k * 8 + j] = X[p * 8 + j]; X[p * 8 + j] = tX;
      }
    float2 piv = A[k * 8 + k];
    for (int j = 0; j < 8; ++j) {
      A[k * 8 + j] = cdiv(A[k * 8 + j], piv);
      X[k * 8 + j] = cdiv(X[k * 8 + j], piv);
    }
    for (int i = 0; i < 8; ++i) {
      if (i == k) continue;
      float2 fct = A[i * 8 + k];
      for (int j = 0; j < 8; ++j) {
        A[i * 8 + j] = csub(A[i * 8 + j], cmul(fct, A[k * 8 + j]));
        X[i * 8 + j] = csub(X[i * 8 + j], cmul(fct, X[k * 8 + j]));
      }
    }
  }
  float2 tr = make_float2(1e-8f, 0.f);               // trace + 1e-8
  for (int c = 0; c < 8; ++c) { tr.x += X[c * 8 + c].x; tr.y += X[c * 8 + c].y; }
  for (int c = 0; c < 8; ++c)
    wvec[(size_t)sys * 8 + c] = cdiv(X[c * 8 + 0], tr);  // ws[:, 0]
}

// ---------------- beamform: enh = conj(w) . spec -> f16 rows for iDFT GEMM ----------------
__global__ void beamform_kernel(const float* __restrict__ spec_ri, const float2* __restrict__ wvec,
                                h16* __restrict__ enh_h) {
  int t = blockIdx.x, b = blockIdx.y;
  int tid = threadIdx.x;
  size_t m = (size_t)b * kT + t;
  for (int f = tid; f < kF; f += 256) {
    float sr_ = 0.f, si_ = 0.f;
    for (int c = 0; c < 8; ++c) {
      size_t r = (size_t)t * kNSEQ + b * 8 + c;
      float sr = spec_ri[r * kDF + f], si = spec_ri[r * kDF + kF + f];
      float2 w = wvec[((size_t)b * kF + f) * 8 + c];
      sr_ += w.x * sr + w.y * si;
      si_ += w.x * si - w.y * sr;
    }
    enh_h[m * kKF + f]      = (h16)sr_;
    enh_h[m * kKF + kF + f] = (h16)si_;
  }
  if (tid < kKF - kDF) enh_h[m * kKF + kDF + tid] = (h16)0.f;
}

// ---------------- iSTFT as WMMA GEMM: frames2 = enh @ idftB^T  [4808 x 512] ----------------
__global__ void gemm_istft(const h16* __restrict__ enh_h, const h16* __restrict__ idftB,
                           float* __restrict__ frames2) {
  int lane = threadIdx.x & 31;
  int wave = threadIdx.x >> 5;
  int mt = blockIdx.x;                             // 0..300
  int nt = blockIdx.y * 8 + wave;                  // 0..31
  int row0 = mt * 16, col0 = nt * 16;
  v8f acc = {};
#pragma unroll
  for (int k0 = 0; k0 < kKF; k0 += 32) {
    v16h a = load_a16(enh_h, kKF, row0, k0, lane);
    v16h b = load_b16(idftB, kKF, col0, k0, lane);
    acc = __builtin_amdgcn_wmma_f32_16x16x32_f16(false, a, false, b, (short)0, acc, false, false);
  }
  int col = col0 + (lane & 15);
  int rb = row0 + ((lane >> 4) << 3);
#pragma unroll
  for (int j = 0; j < 8; ++j)
    if (rb + j < kB * kT)
      frames2[(size_t)(rb + j) * kNFFT + col] = acc[j];
}

// ---------------- windowed overlap-add ----------------
__global__ void ola_kernel(const float* __restrict__ frames2, float* __restrict__ acc) {
  int t = blockIdx.x, b = blockIdx.y;
  int tid = threadIdx.x;
  size_t m = (size_t)b * kT + t;
  for (int k = tid; k < kNFFT; k += 256) {
    float v = frames2[m * kNFFT + k] * hann_win(k);
    atomicAdd(&acc[(size_t)b * kTOTAL + t * kHOP + k], v);
  }
}

__global__ void wsq_kernel(float* __restrict__ wsq) {
  int t = blockIdx.x, tid = threadIdx.x;
  for (int k = tid; k < kNFFT; k += 256) {
    float w = hann_win(k);
    atomicAdd(&wsq[t * kHOP + k], w * w);
  }
}

__global__ void final_kernel(const float* __restrict__ acc, const float* __restrict__ wsq,
                             float* __restrict__ out) {
  size_t idx = (size_t)blockIdx.x * blockDim.x + threadIdx.x;
  if (idx >= (size_t)kB * kL) return;
  int b = (int)(idx / kL);
  int l = (int)(idx - (size_t)b * kL);
  out[idx] = acc[(size_t)b * kTOTAL + kPAD + l] / fmaxf(wsq[kPAD + l], 1e-11f);
}

// ---------------- host-side launch ----------------
extern "C" void kernel_launch(void* const* d_in, const int* in_sizes, int n_in,
                              void* d_out, int out_size, void* d_ws, size_t ws_size,
                              hipStream_t stream) {
  (void)in_sizes; (void)n_in; (void)out_size; (void)ws_size;
  const float* x   = (const float*)d_in[0];
  const float* Wih = (const float*)d_in[1];
  const float* Whh = (const float*)d_in[2];
  const float* bih = (const float*)d_in[3];
  const float* bhh = (const float*)d_in[4];
  const float* W1  = (const float*)d_in[5];
  const float* b1  = (const float*)d_in[6];
  const float* W2  = (const float*)d_in[7];
  const float* b2  = (const float*)d_in[8];

  char* ws = (char*)d_ws;
  size_t off = 0;
  auto alloc = [&](size_t bytes) -> char* {
    char* p = ws + off;
    off = (off + bytes + 255) & ~(size_t)255;
    return p;
  };
  h16*    frames = (h16*)   alloc((size_t)kM * kNFFT * 2);
  h16*    feat_h = (h16*)   alloc((size_t)kM * kKF * 2);
  float*  spec   = (float*) alloc((size_t)kM * kDF * 4);
  h16*    dftB   = (h16*)   alloc((size_t)kNPAD * kNFFT * 2);
  h16*    idftB  = (h16*)   alloc((size_t)kNFFT * kKF * 2);
  h16*    Wih_h  = (h16*)   alloc((size_t)kG * kKF * 2);
  h16*    Whh_h  = (h16*)   alloc((size_t)kG * kH * 2);
  h16*    W1h    = (h16*)   alloc((size_t)kNPAD * kH * 2);
  h16*    W2h    = (h16*)   alloc((size_t)kNPAD * kH * 2);
  float*  bias   = (float*) alloc((size_t)kG * 4);
  float*  xW     = (float*) alloc((size_t)kM * kG * 4);
  h16*    h_all  = (h16*)   alloc((size_t)kM * kH * 2);
  float*  s_ri   = (float*) alloc((size_t)kM * kDF * 4);
  float*  n_ri   = (float*) alloc((size_t)kM * kDF * 4);
  float2* psd_s  = (float2*)alloc((size_t)kB * kF * 64 * 8);
  float2* psd_n  = (float2*)alloc((size_t)kB * kF * 64 * 8);
  float2* wvec   = (float2*)alloc((size_t)kB * kF * 8 * 8);
  h16*    enh_h  = (h16*)   alloc((size_t)kMI * kKF * 2);
  float*  frm2   = (float*) alloc((size_t)kMI * kNFFT * 4);
  float*  accb   = (float*) alloc((size_t)kB * kTOTAL * 4);
  float*  wsq    = (float*) alloc((size_t)kTOTAL * 4);

  (void)hipMemsetAsync(accb, 0, (size_t)kB * kTOTAL * 4, stream);
  (void)hipMemsetAsync(wsq, 0, (size_t)kTOTAL * 4, stream);
  (void)hipMemsetAsync(enh_h, 0, (size_t)kMI * kKF * 2, stream);

  prep_weights<<<256, 256, 0, stream>>>(Wih, Whh, bih, bhh, W1, W2,
                                        Wih_h, Whh_h, W1h, W2h, bias);
  prep_basis<<<512, 256, 0, stream>>>(dftB, idftB);
  frames_kernel<<<dim3(kT, kNSEQ), 256, 0, stream>>>(x, frames);
  gemm_spec<<<dim3(kM / 16, 5), 256, 0, stream>>>(frames, dftB, spec, feat_h);
  {
    size_t tot = (size_t)kM * (kKF - kDF);
    feat_pad<<<(unsigned)((tot + 255) / 256), 256, 0, stream>>>(feat_h);
  }
  gemm_xw<<<dim3(kM / 16, 4), 256, 0, stream>>>(feat_h, Wih_h, bias, xW);
  lstm_kernel<<<1, 1024, 311296, stream>>>(xW, Whh_h, h_all);
  gemm_mask<<<dim3(kM / 16, 5), 256, 0, stream>>>(h_all, W1h, b1, s_ri);
  gemm_mask<<<dim3(kM / 16, 5), 256, 0, stream>>>(h_all, W2h, b2, n_ri);
  {
    size_t tot = (size_t)kM * kF;
    mask_apply<<<(unsigned)((tot + 255) / 256), 256, 0, stream>>>(s_ri, n_ri, spec);
  }
  psd_kernel<<<dim3(kF, kB), 64, 0, stream>>>(s_ri, n_ri, psd_s, psd_n);
  solve_kernel<<<(kB * kF + 31) / 32, 32, 0, stream>>>(psd_s, psd_n, wvec);
  beamform_kernel<<<dim3(kT, kB), 256, 0, stream>>>(spec, wvec, enh_h);
  gemm_istft<<<dim3(kMI / 16, 4), 256, 0, stream>>>(enh_h, idftB, frm2);
  wsq_kernel<<<kT, 256, 0, stream>>>(wsq);
  ola_kernel<<<dim3(kT, kB), 256, 0, stream>>>(frm2, accb);
  {
    size_t tot = (size_t)kB * kL;
    final_kernel<<<(unsigned)((tot + 255) / 256), 256, 0, stream>>>(accb, wsq, (float*)d_out);
  }
}
